// MultiHeadAttention_47201690583727
// MI455X (gfx1250) — matrix operand order repaired and checked by reading, not compiled, not run on hardware
//
#include <hip/hip_runtime.h>

typedef __attribute__((ext_vector_type(16))) _Float16 v16h;
typedef __attribute__((ext_vector_type(8)))  _Float16 v8h;
typedef __attribute__((ext_vector_type(8)))  float    v8f;

#define D_MODEL 1024
#define SEQ_N   2048
#define NHEAD   16
#define DHEAD   64

union FragA { v16h f; v8h h[2]; };

static __device__ __forceinline__ v8f wmma_f16(v16h a, v16h b, v8f c) {
  // (neg_a, A, neg_b, B, c_mod, C, reuse_a, reuse_b) -> v_wmma_f32_16x16x32_f16
  return __builtin_amdgcn_wmma_f32_16x16x32_f16(false, a, false, b, (short)0, c,
                                                false, false);
}

// ---------------------------------------------------------------------------
// Kernel 0: convert weights f32 -> f16 (weights then stay resident in L2)
// ---------------------------------------------------------------------------
__global__ void convert_weights(const float* __restrict__ wq,
                                const float* __restrict__ wo,
                                _Float16* __restrict__ wq_h,
                                _Float16* __restrict__ wo_h) {
  int i = blockIdx.x * 256 + threadIdx.x;  // grid covers 1024*1024
  wq_h[i] = (_Float16)wq[i];
  wo_h[i] = (_Float16)wo[i];
}

// ---------------------------------------------------------------------------
// GEMM: Y(m,col) = sum_k A(m,k) * W(col,k) + bias[col]
// MODE 0: A = one of {Q,K,V} (f32, converted on load), out -> head-major f16
// MODE 1: A = ctx (f16), out -> f32 d_out
// Block: 256 thr = 8 waves (2x4); tile M=32, N=128; K staged 64 at a time.
// ---------------------------------------------------------------------------
template<int MODE>
__global__ void __launch_bounds__(256, 1)
gemm_proj(const float* __restrict__ Xq, const float* __restrict__ Xk,
          const float* __restrict__ Xv, const _Float16* __restrict__ ctx,
          const _Float16* __restrict__ W, const float* __restrict__ bias,
          _Float16* __restrict__ oq, _Float16* __restrict__ ok,
          _Float16* __restrict__ ov, float* __restrict__ outF) {
  __shared__ __align__(16) _Float16 Ash[32][72];   // padded stride: conflict-free

  const int tid  = threadIdx.x;
  const int lane = tid & 31;
  const int wid  = tid >> 5;
  const int wrow = wid >> 2;            // 0..1
  const int wcol = wid & 3;             // 0..3
  const int lmod  = lane & 15;
  const int lhalf = (lane < 16) ? 0 : 1;

  const int mbase   = blockIdx.y * 32;
  const int colbase = blockIdx.x * 128 + wcol * 32;

  const float* src = nullptr;
  if (MODE == 0)
    src = (blockIdx.z == 0) ? Xq : ((blockIdx.z == 1) ? Xk : Xv);

  v8f acc0 = {0.f,0.f,0.f,0.f,0.f,0.f,0.f,0.f};
  v8f acc1 = acc0;

  const int arow  = tid >> 3;           // 0..31
  const int acol8 = (tid & 7) * 8;      // 0..56

  for (int kc = 0; kc < D_MODEL; kc += 64) {
    __syncthreads();
    if (MODE == 0) {
      const float* p = src + (size_t)(mbase + arow) * D_MODEL + kc + acol8;
      v8h tmp;
      #pragma unroll
      for (int j = 0; j < 8; ++j) tmp[j] = (_Float16)p[j];
      *(v8h*)&Ash[arow][acol8] = tmp;
    } else {
      const _Float16* p = ctx + (size_t)(mbase + arow) * D_MODEL + kc + acol8;
      *(v8h*)&Ash[arow][acol8] = *(const v8h*)p;
    }
    __syncthreads();

    #pragma unroll
    for (int kk = 0; kk < 64; kk += 32) {
      FragA a;
      a.h[0] = *(const v8h*)&Ash[wrow * 16 + lmod][kk + lhalf * 16];
      a.h[1] = *(const v8h*)&Ash[wrow * 16 + lmod][kk + 8 + lhalf * 16];
      // B fragment: lane holds fixed column (weight row), contiguous K run
      const _Float16* w0 = W + (size_t)(colbase + 0  + lmod) * D_MODEL + kc + kk + lhalf * 16;
      const _Float16* w1 = W + (size_t)(colbase + 16 + lmod) * D_MODEL + kc + kk + lhalf * 16;
      FragA b0, b1;
      b0.h[0] = *(const v8h*)(w0);     b0.h[1] = *(const v8h*)(w0 + 8);
      b1.h[0] = *(const v8h*)(w1);     b1.h[1] = *(const v8h*)(w1 + 8);
      acc0 = wmma_f16(a.f, b0.f, acc0);
      acc1 = wmma_f16(a.f, b1.f, acc1);
    }
  }

  #pragma unroll
  for (int nt = 0; nt < 2; ++nt) {
    v8f acc = nt ? acc1 : acc0;
    const int col = colbase + nt * 16 + lmod;
    const float bv = bias[col];
    #pragma unroll
    for (int r = 0; r < 8; ++r) {
      const int mg = mbase + wrow * 16 + r + lhalf * 8;   // C layout: M = r (+8 hi half)
      const float val = acc[r] + bv;
      if (MODE == 0) {
        const int b  = mg >> 11;            // /2048
        const int ns = mg & 2047;
        const int h  = col >> 6;
        const int dv = col & 63;
        _Float16* dst = (blockIdx.z == 0) ? oq : ((blockIdx.z == 1) ? ok : ov);
        dst[(((size_t)b * NHEAD + h) * SEQ_N + ns) * DHEAD + dv] = (_Float16)val;
      } else {
        outF[(size_t)mg * D_MODEL + col] = val;
      }
    }
  }
}

// ---------------------------------------------------------------------------
// Flash attention (causal, online softmax). Block = 128 thr = 4 waves.
// Each block: one (b,h) and 64 query rows (16 per wave). Key blocks of 64.
// __launch_bounds__(128, 1): relax VGPR budget -> no scratch spills in the
// WMMA inner loop (round-1 asm showed Folded Reload scratch traffic).
// ---------------------------------------------------------------------------
__global__ void __launch_bounds__(128, 1)
flash_attn(const _Float16* __restrict__ qh,
           const _Float16* __restrict__ kh,
           const _Float16* __restrict__ vh,
           _Float16* __restrict__ ctx) {
  __shared__ __align__(16) _Float16 Ksh [64][72];       // [key][d]   row-major
  __shared__ __align__(16) _Float16 VshT[64][72];       // [d][key]   transposed
  __shared__ __align__(16) _Float16 Psh [4][16][72];    // per-wave P tile

  const int tid   = threadIdx.x;
  const int lane  = tid & 31;
  const int wid   = tid >> 5;
  const int lmod  = lane & 15;
  const int lhalf = (lane < 16) ? 0 : 1;

  const int qblock = blockIdx.x;            // 0..31
  const int bh     = blockIdx.y;            // 0..31
  const int qbase  = qblock * 64;
  const size_t base = (size_t)bh * SEQ_N * DHEAD;

  // Q fragments for this wave's 16 rows, kept in registers for all key blocks
  FragA aq[2];
  {
    const _Float16* qp = qh + base + (size_t)(qbase + wid * 16 + lmod) * DHEAD;
    #pragma unroll
    for (int kc = 0; kc < 2; ++kc) {
      aq[kc].h[0] = *(const v8h*)(qp + kc * 32 + lhalf * 16);
      aq[kc].h[1] = *(const v8h*)(qp + kc * 32 + 8 + lhalf * 16);
    }
  }

  float mrow[8], lrow[8];
  v8f oacc[4];
  #pragma unroll
  for (int r = 0; r < 8; ++r) { mrow[r] = -3.0e38f; lrow[r] = 0.f; }
  #pragma unroll
  for (int t = 0; t < 4; ++t) {
    v8f z = {0.f,0.f,0.f,0.f,0.f,0.f,0.f,0.f};
    oacc[t] = z;
  }

  const float scale = 0.125f;   // 1/sqrt(64)
  const int qi_lane = qbase + wid * 16 + lhalf * 8;   // + r gives this lane's row

  for (int kb = 0; kb <= qblock; ++kb) {
    const int kbase = kb * 64;
    __syncthreads();
    // cooperative load: K row-major, V transposed
    #pragma unroll
    for (int i = 0; i < 4; ++i) {
      const int chunk = tid + i * 128;                 // 0..511
      const int row = chunk >> 3;
      const int cg  = (chunk & 7) * 8;
      v8h kv = *(const v8h*)(kh + base + (size_t)(kbase + row) * DHEAD + cg);
      *(v8h*)&Ksh[row][cg] = kv;
      v8h vv = *(const v8h*)(vh + base + (size_t)(kbase + row) * DHEAD + cg);
      #pragma unroll
      for (int j = 0; j < 8; ++j) VshT[cg + j][row] = vv[j];
    }
    __syncthreads();

    // S = Q * K^T  (4 column tiles of 16 keys, K-dim 64 = 2 WMMAs each)
    v8f st[4];
    #pragma unroll
    for (int t = 0; t < 4; ++t) {
      v8f z = {0.f,0.f,0.f,0.f,0.f,0.f,0.f,0.f};
      FragA b0, b1;
      b0.h[0] = *(const v8h*)&Ksh[t * 16 + lmod][0  + lhalf * 16];
      b0.h[1] = *(const v8h*)&Ksh[t * 16 + lmod][8  + lhalf * 16];
      b1.h[0] = *(const v8h*)&Ksh[t * 16 + lmod][32 + lhalf * 16];
      b1.h[1] = *(const v8h*)&Ksh[t * 16 + lmod][40 + lhalf * 16];
      z = wmma_f16(aq[0].f, b0.f, z);
      z = wmma_f16(aq[1].f, b1.f, z);
      st[t] = z;
    }

    // scale + causal mask (diagonal block only)
    const bool diag = (kb == qblock);
    #pragma unroll
    for (int t = 0; t < 4; ++t) {
      const int kj = kbase + t * 16 + lmod;
      #pragma unroll
      for (int r = 0; r < 8; ++r) {
        float s = st[t][r] * scale;
        if (diag && (kj > qi_lane + r)) s = -3.0e38f;
        st[t][r] = s;
      }
    }

    // online softmax: row reductions across 16 lanes of each half-wave
    #pragma unroll
    for (int r = 0; r < 8; ++r) {
      float mx = fmaxf(fmaxf(st[0][r], st[1][r]), fmaxf(st[2][r], st[3][r]));
      #pragma unroll
      for (int off = 1; off < 16; off <<= 1)
        mx = fmaxf(mx, __shfl_xor(mx, off, 32));
      const float mnew = fmaxf(mrow[r], mx);
      const float corr = __expf(mrow[r] - mnew);
      mrow[r] = mnew;
      float rs = 0.f;
      #pragma unroll
      for (int t = 0; t < 4; ++t) {
        const float p = __expf(st[t][r] - mnew);
        st[t][r] = p;
        rs += p;
      }
      #pragma unroll
      for (int off = 1; off < 16; off <<= 1)
        rs += __shfl_xor(rs, off, 32);
      lrow[r] = lrow[r] * corr + rs;
      #pragma unroll
      for (int t = 0; t < 4; ++t) oacc[t][r] *= corr;
    }

    // P: C-layout -> LDS (per-wave region, no barrier needed)
    #pragma unroll
    for (int t = 0; t < 4; ++t)
      #pragma unroll
      for (int r = 0; r < 8; ++r)
        Psh[wid][r + lhalf * 8][t * 16 + lmod] = (_Float16)st[t][r];

    // O += P * V   (A from Psh, B from transposed V)
    #pragma unroll
    for (int kc = 0; kc < 2; ++kc) {
      FragA pa;
      pa.h[0] = *(const v8h*)&Psh[wid][lmod][kc * 32 + lhalf * 16];
      pa.h[1] = *(const v8h*)&Psh[wid][lmod][kc * 32 + 8 + lhalf * 16];
      #pragma unroll
      for (int t = 0; t < 4; ++t) {
        FragA bv;
        bv.h[0] = *(const v8h*)&VshT[t * 16 + lmod][kc * 32 + lhalf * 16];
        bv.h[1] = *(const v8h*)&VshT[t * 16 + lmod][kc * 32 + 8 + lhalf * 16];
        oacc[t] = wmma_f16(pa.f, bv.f, oacc[t]);
      }
    }
  }

  // normalize + store ctx back to [B, N, d_model] (f16) for output projection
  const int b = bh >> 4, h = bh & 15;
  #pragma unroll
  for (int r = 0; r < 8; ++r) {
    const float inv = 1.0f / lrow[r];
    const int ns = qbase + wid * 16 + r + lhalf * 8;
    #pragma unroll
    for (int t = 0; t < 4; ++t) {
      const int dv = t * 16 + lmod;
      ctx[((size_t)b * SEQ_N + ns) * D_MODEL + h * DHEAD + dv] =
          (_Float16)(oacc[t][r] * inv);
    }
  }
}

// ---------------------------------------------------------------------------
extern "C" void kernel_launch(void* const* d_in, const int* in_sizes, int n_in,
                              void* d_out, int out_size, void* d_ws, size_t ws_size,
                              hipStream_t stream) {
  (void)in_sizes; (void)n_in; (void)out_size; (void)ws_size;

  const float* Q  = (const float*)d_in[0];
  const float* K  = (const float*)d_in[1];
  const float* V  = (const float*)d_in[2];
  const float* wq = (const float*)d_in[3];
  const float* bq = (const float*)d_in[4];
  const float* wo = (const float*)d_in[5];
  const float* bo = (const float*)d_in[6];
  float* out = (float*)d_out;

  // workspace layout (36 MB total)
  char* ws = (char*)d_ws;
  _Float16* wq_h = (_Float16*)(ws);                       //  2 MB
  _Float16* wo_h = (_Float16*)(ws + (size_t)( 2u << 20)); //  2 MB
  _Float16* q_h  = (_Float16*)(ws + (size_t)( 4u << 20)); //  8 MB [B,H,N,64]
  _Float16* k_h  = (_Float16*)(ws + (size_t)(12u << 20)); //  8 MB
  _Float16* v_h  = (_Float16*)(ws + (size_t)(20u << 20)); //  8 MB
  _Float16* ctx  = (_Float16*)(ws + (size_t)(28u << 20)); //  8 MB [B,N,1024]

  convert_weights<<<dim3((D_MODEL * D_MODEL) / 256), 256, 0, stream>>>(wq, wo, wq_h, wo_h);

  // fused Q/K/V projection: z selects the source matrix
  gemm_proj<0><<<dim3(D_MODEL / 128, (2 * SEQ_N) / 32, 3), 256, 0, stream>>>(
      Q, K, V, nullptr, wq_h, bq, q_h, k_h, v_h, nullptr);

  flash_attn<<<dim3(SEQ_N / 64, 2 * NHEAD), 128, 0, stream>>>(q_h, k_h, v_h, ctx);

  gemm_proj<1><<<dim3(D_MODEL / 128, (2 * SEQ_N) / 32, 1), 256, 0, stream>>>(
      nullptr, nullptr, nullptr, ctx, wo_h, bo, nullptr, nullptr, nullptr, out);
}
